// LGF_Mamba_88467736363572
// MI455X (gfx1250) — matrix operand order repaired and checked by reading, not compiled, not run on hardware
//
#include <hip/hip_runtime.h>
#include <hip/hip_bf16.h>

// ---------------- constants (match reference) ----------------
#define BB   16
#define LL   1024
#define DM   96
#define DSN  16     // state dim
#define DC   4      // causal conv width
#define DI   192
#define D2   384    // 2*DI
#define RR   6      // dt rank
#define CI   32     // channel-scan channels (== H == W)
#define HH   32
#define NJ   38     // RR + 2*DSN
#define LCH  12288  // D2 * HH
#define SMAXP 1536  // max C*S over branches (192*8 == 32*48)

#ifndef __has_builtin
#define __has_builtin(x) 0
#endif
#if __has_builtin(__builtin_amdgcn_global_load_async_to_lds_b128) && \
    __has_builtin(__builtin_amdgcn_s_wait_asynccnt)
#define USE_ASYNC_LDS 1
#else
#define USE_ASYNC_LDS 0
#endif

typedef __attribute__((ext_vector_type(16))) _Float16 v16h;
typedef __attribute__((ext_vector_type(8)))  float    v8f;

__device__ __forceinline__ float dev_rcp(float x) { return __builtin_amdgcn_rcpf(x); }
__device__ __forceinline__ float dev_silu(float x) { return x * dev_rcp(1.f + __expf(-x)); }
__device__ __forceinline__ float dev_softplus(float x) {
  return (x > 20.f) ? x : __logf(1.f + __expf(x));
}

// ---------------- depthwise 3x3 conv on 32x32 grid, channel-last layout -----
__global__ void k_dwconv3x3(const float* __restrict__ in, const float* __restrict__ w,
                            const float* __restrict__ bias, float* __restrict__ out, int C) {
  int idx = blockIdx.x * blockDim.x + threadIdx.x;
  int total = BB * LL * C;
  if (idx >= total) return;
  int c = idx % C;
  int l = (idx / C) % LL;
  int b = idx / (C * LL);
  int y = l >> 5, x = l & 31;
  float s = bias[c];
#pragma unroll
  for (int dy = -1; dy <= 1; ++dy) {
    int yy = y + dy;
    if (yy < 0 || yy >= HH) continue;
#pragma unroll
    for (int dx = -1; dx <= 1; ++dx) {
      int xx = x + dx;
      if (xx < 0 || xx >= HH) continue;
      s += w[c * 9 + (dy + 1) * 3 + (dx + 1)] *
           in[((size_t)b * LL + yy * 32 + xx) * C + c];
    }
  }
  out[idx] = s;
}

// ---------------- WMMA GEMM:  Y[M,N] = act( X[M,K] * W[N,K]^T + bias ) ------
// X row-major (stride K), W row-major (stride K).  K in {32,96,192} (mult of 32),
// M multiple of 16.  Block = 128 threads (4 wave32), covers 16 rows x 64 cols.
// Stage f32 tiles via CDNA5 async-to-LDS, convert once to f16 in LDS, then run
// the tight ds_load_b128 -> v_wmma_f32_16x16x32_f16 loop.
template <int ACT>   // 0 = none, 1 = SiLU
__global__ void k_gemm_wmma(const float* __restrict__ X, const float* __restrict__ W,
                            const float* __restrict__ bias, float* __restrict__ Y,
                            int N, int K) {
  __shared__ __align__(16) float    sXf[16 * 192];
  __shared__ __align__(16) float    sWf[64 * 192];
  __shared__ __align__(16) _Float16 sXh[16 * 192];
  __shared__ __align__(16) _Float16 sWh[64 * 192];
  const int tid = threadIdx.x;
  const size_t row0 = (size_t)blockIdx.x * 16;
  const int n0 = blockIdx.y * 64;
  const int wvalid = (N - n0 < 64 ? N - n0 : 64) * K;   // multiple of 4 (K mult of 32)

#if USE_ASYNC_LDS
  {
    typedef int v4i __attribute__((vector_size(16)));
    typedef __attribute__((address_space(1))) v4i gv4i;
    typedef __attribute__((address_space(3))) v4i lv4i;
    const float* gX = X + row0 * K;
    const float* gW = W + (size_t)n0 * K;
    for (int i4 = tid * 4; i4 < 16 * K; i4 += 128 * 4)
      __builtin_amdgcn_global_load_async_to_lds_b128(
          (gv4i*)(gX + i4), (lv4i*)(sXf + i4), 0, 0);
    for (int i4 = tid * 4; i4 < wvalid; i4 += 128 * 4)
      __builtin_amdgcn_global_load_async_to_lds_b128(
          (gv4i*)(gW + i4), (lv4i*)(sWf + i4), 0, 0);
    __builtin_amdgcn_s_wait_asynccnt(0);
  }
#else
  for (int i = tid; i < 16 * K; i += 128) sXf[i] = X[row0 * K + i];
  for (int i = tid; i < wvalid; i += 128) sWf[i] = W[(size_t)n0 * K + i];
#endif
  __syncthreads();
  // one-shot cooperative f32 -> f16 conversion (amortized across all 4 waves)
  for (int i = tid; i < 16 * K; i += 128) sXh[i] = (_Float16)sXf[i];
  for (int i = tid; i < 64 * K; i += 128)
    sWh[i] = (i < wvalid) ? (_Float16)sWf[i] : (_Float16)0.f;
  __syncthreads();

  const int wv = tid >> 5;      // wave id: owns N sub-tile [wv*16, wv*16+16)
  const int lid = tid & 31;
  const int m16 = lid & 15;     // row (A) / col (B) index within tile
  const int half = lid >> 4;    // lane half selects K sub-range

  v8f c = {};
  for (int kc = 0; kc < K; kc += 32) {
    v16h a, bm;
#pragma unroll
    for (int v = 0; v < 8; ++v) {
      int grp = v >> 2, vv = v & 3;
      int ka = kc + grp * 16 + half * 8 + vv * 2;   // A 16x32 f16 layout
      a[2 * v]     = sXh[m16 * K + ka];
      a[2 * v + 1] = sXh[m16 * K + ka + 1];
      int kb = kc + grp * 16 + half * 8 + vv * 2;   // B 32x16 f16 layout
      bm[2 * v]     = sWh[(wv * 16 + m16) * K + kb];
      bm[2 * v + 1] = sWh[(wv * 16 + m16) * K + kb + 1];
    }
    c = __builtin_amdgcn_wmma_f32_16x16x32_f16(false, a, false, bm,
                                               (short)0, c, false, false);
  }

#pragma unroll
  for (int r = 0; r < 8; ++r) {               // C/D: m = half*8 + r, n = lid&15
    int m = half * 8 + r;
    int col = n0 + wv * 16 + m16;
    if (col < N) {
      float v = c[r] + (bias ? bias[col] : 0.f);
      if (ACT) v = dev_silu(v);
      Y[(row0 + m) * (size_t)N + col] = v;
    }
  }
}

// ---------------- branch input gather (virtual u0) --------------------------
__device__ __forceinline__ float load_u0(const float* __restrict__ xz, int mode,
                                         int rev, int b, int c, int t, int Lb) {
  if (t < 0) return 0.f;   // causal zero pad
  int tt = rev ? (Lb - 1 - t) : t;
  size_t idx;
  if (mode == 0) {
    idx = ((size_t)b * LL + tt) * D2 + c;
  } else if (mode == 1) {
    int c384 = tt >> 5, w = tt & 31;
    idx = ((size_t)b * LL + c * 32 + w) * D2 + c384;
  } else {
    int c384 = tt >> 5, h = tt & 31;
    idx = ((size_t)b * LL + h * 32 + c) * D2 + c384;
  }
  return xz[idx];
}

// depthwise causal conv1d (k=4) + SiLU -> xbuf (B, Lb, C) in scan order
__global__ void k_causal_conv(const float* __restrict__ xz, const float* __restrict__ cw,
                              const float* __restrict__ cb, float* __restrict__ xbuf,
                              int mode, int rev, int C, int Lb) {
  size_t idx = (size_t)blockIdx.x * blockDim.x + threadIdx.x;
  size_t total = (size_t)BB * Lb * C;
  if (idx >= total) return;
  int c = (int)(idx % C);
  int t = (int)((idx / C) % Lb);
  int b = (int)(idx / ((size_t)C * Lb));
  float s = cb[c];
#pragma unroll
  for (int k = 0; k < DC; ++k)
    s += cw[c * DC + k] * load_u0(xz, mode, rev, b, c, t - (DC - 1) + k, Lb);
  xbuf[idx] = dev_silu(s);
}

// ---------------- chunked selective scan ------------------------------------
// h_t[n] = exp(delta_t*A[n])*h_{t-1}[n] + delta_t*u_t*B_t[n] is diagonal, so a
// chunk transition collapses to h_end = exp(A[n]*sum(delta)) * h_start + r.
// Pass A: per-chunk (sumdelta, r) with h_start = 0.    (parallel, B*C*S threads)
// Pass B: stitch chunk-initial states sequentially.    (B*C threads, S steps)
// Pass C: rescan each chunk from its true initial state, emit y into acc.

__global__ void k_scan_chunkA(const float* __restrict__ xbuf, const float* __restrict__ xdbl,
                              const float* __restrict__ A_log, const float* __restrict__ dtw,
                              const float* __restrict__ dtb, float* __restrict__ cbuf,
                              int C, int Lb, int S, int CL) {
  int idx = blockIdx.x * blockDim.x + threadIdx.x;
  if (idx >= BB * C * S) return;
  int c = idx % C;
  int s = (idx / C) % S;
  int b = idx / (C * S);

  float A[DSN], h[DSN];
#pragma unroll
  for (int n = 0; n < DSN; ++n) { A[n] = -__expf(A_log[c * DSN + n]); h[n] = 0.f; }
  float w0 = dtw[c * RR + 0], w1 = dtw[c * RR + 1], w2 = dtw[c * RR + 2];
  float w3 = dtw[c * RR + 3], w4 = dtw[c * RR + 4], w5 = dtw[c * RR + 5];
  const float bdt = dtb[c];

  float sd = 0.f;
  const int t0 = s * CL, t1 = t0 + CL;
  for (int t = t0; t < t1; ++t) {
    const float* xd = xdbl + ((size_t)b * Lb + t) * NJ;
    if (t + 16 < Lb) {
      __builtin_prefetch(xdbl + ((size_t)b * Lb + t + 16) * NJ, 0, 1);
      __builtin_prefetch(xbuf + ((size_t)b * Lb + t + 16) * C + c, 0, 1);
    }
    float dt = bdt + w0 * xd[0] + w1 * xd[1] + w2 * xd[2] +
                     w3 * xd[3] + w4 * xd[4] + w5 * xd[5];
    float delta = dev_softplus(dt);
    sd += delta;
    float du = delta * xbuf[((size_t)b * Lb + t) * C + c];
#pragma unroll
    for (int n = 0; n < DSN; ++n)
      h[n] = h[n] * __expf(delta * A[n]) + du * xd[RR + n];
  }
  float* cb = cbuf + ((size_t)(b * C + c) * S + s) * 17;
  cb[0] = sd;
#pragma unroll
  for (int n = 0; n < DSN; ++n) cb[1 + n] = h[n];
}

__global__ void k_scan_stitch(const float* __restrict__ cbuf, const float* __restrict__ A_log,
                              float* __restrict__ hinit, int C, int S) {
  int idx = blockIdx.x * blockDim.x + threadIdx.x;
  if (idx >= BB * C) return;
  int c = idx % C, b = idx / C;
  float A[DSN], h[DSN];
#pragma unroll
  for (int n = 0; n < DSN; ++n) { A[n] = -__expf(A_log[c * DSN + n]); h[n] = 0.f; }
  for (int s = 0; s < S; ++s) {
    float* hi = hinit + ((size_t)(b * C + c) * S + s) * DSN;
    const float* cb = cbuf + ((size_t)(b * C + c) * S + s) * 17;
    float sd = cb[0];
#pragma unroll
    for (int n = 0; n < DSN; ++n) {
      hi[n] = h[n];
      h[n] = __expf(A[n] * sd) * h[n] + cb[1 + n];
    }
  }
}

__global__ void k_scan_chunkC(const float* __restrict__ xbuf, const float* __restrict__ xdbl,
                              const float* __restrict__ xz, const float* __restrict__ A_log,
                              const float* __restrict__ Dw, const float* __restrict__ dtw,
                              const float* __restrict__ dtb, const float* __restrict__ hinit,
                              float* __restrict__ acc,
                              int mode, int rev, int C, int Lb, int S, int CL) {
  int idx = blockIdx.x * blockDim.x + threadIdx.x;
  if (idx >= BB * C * S) return;
  int c = idx % C;
  int s = (idx / C) % S;
  int b = idx / (C * S);

  float A[DSN], h[DSN];
  const float* hi = hinit + ((size_t)(b * C + c) * S + s) * DSN;
#pragma unroll
  for (int n = 0; n < DSN; ++n) { A[n] = -__expf(A_log[c * DSN + n]); h[n] = hi[n]; }
  float w0 = dtw[c * RR + 0], w1 = dtw[c * RR + 1], w2 = dtw[c * RR + 2];
  float w3 = dtw[c * RR + 3], w4 = dtw[c * RR + 4], w5 = dtw[c * RR + 5];
  const float bdt = dtb[c], Dv = Dw[c];

  const int t0 = s * CL, t1 = t0 + CL;
  for (int t = t0; t < t1; ++t) {
    const float* xd = xdbl + ((size_t)b * Lb + t) * NJ;
    if (t + 16 < Lb) {
      __builtin_prefetch(xdbl + ((size_t)b * Lb + t + 16) * NJ, 0, 1);
      __builtin_prefetch(xbuf + ((size_t)b * Lb + t + 16) * C + c, 0, 1);
    }
    float dt = bdt + w0 * xd[0] + w1 * xd[1] + w2 * xd[2] +
                     w3 * xd[3] + w4 * xd[4] + w5 * xd[5];
    float delta = dev_softplus(dt);
    float u = xbuf[((size_t)b * Lb + t) * C + c];
    float du = delta * u;
    float y = Dv * u;
#pragma unroll
    for (int n = 0; n < DSN; ++n) {
      h[n] = h[n] * __expf(delta * A[n]) + du * xd[RR + n];
      y += h[n] * xd[RR + DSN + n];
    }
    int tt = rev ? (Lb - 1 - t) : t;
    size_t aidx; bool valid = true;
    if (mode == 0) {
      float z = xz[((size_t)b * LL + tt) * D2 + DI + c];
      y *= dev_silu(z);
      aidx = ((size_t)b * LL + tt) * DI + c;
    } else if (mode == 1) {
      int c384 = tt >> 5, w = tt & 31;
      valid = (c384 < DI);
      aidx = ((size_t)b * LL + c * 32 + w) * DI + c384;
    } else {
      int c384 = tt >> 5, hh = tt & 31;
      valid = (c384 < DI);
      aidx = ((size_t)b * LL + hh * 32 + c) * DI + c384;
    }
    if (valid) acc[aidx] += y;   // unique element per kernel; launches serialize
  }
}

// ---------------- finalize: out = acc * out_conv ----------------------------
__global__ void k_finalize(const float* __restrict__ acc, const float* __restrict__ oc,
                           float* __restrict__ out) {
  int idx = blockIdx.x * blockDim.x + threadIdx.x;
  if (idx < BB * LL * DI) out[idx] = acc[idx] * oc[idx];
}

// ---------------------------------------------------------------------------
extern "C" void kernel_launch(void* const* d_in, const int* in_sizes, int n_in,
                              void* d_out, int out_size, void* d_ws, size_t ws_size,
                              hipStream_t stream) {
  (void)in_sizes; (void)n_in; (void)out_size; (void)ws_size;
  const float* hs   = (const float*)d_in[0];
  const float* dw1w = (const float*)d_in[1];
  const float* dw1b = (const float*)d_in[2];
  const float* pw1w = (const float*)d_in[3];
  const float* pw1b = (const float*)d_in[4];
  const float* dw2w = (const float*)d_in[5];
  const float* dw2b = (const float*)d_in[6];
  const float* pw2w = (const float*)d_in[7];
  const float* pw2b = (const float*)d_in[8];
  const float* ipw  = (const float*)d_in[9];
  float* out = (float*)d_out;

  // workspace layout (floats)
  float* acc   = (float*)d_ws;                         // B*L*DI
  float* xz    = acc   + (size_t)BB * LL * DI;         // B*L*D2
  float* t1    = xz    + (size_t)BB * LL * D2;         // B*L*DM
  float* t2    = t1    + (size_t)BB * LL * DM;
  float* t3    = t2    + (size_t)BB * LL * DM;
  float* ocv   = t3    + (size_t)BB * LL * DM;         // B*L*DI
  float* xbuf  = ocv   + (size_t)BB * LL * DI;         // B*LCH*CI (covers f: B*LL*DI)
  float* xdb   = xbuf  + (size_t)BB * LCH * CI;        // B*LCH*NJ
  float* cbuf  = xdb   + (size_t)BB * LCH * NJ;        // B*SMAXP*17
  float* hinit = cbuf  + (size_t)BB * SMAXP * 17;      // B*SMAXP*16

  (void)hipMemsetAsync(acc, 0, (size_t)BB * LL * DI * sizeof(float), stream);

  const int TPB = 256;
  // ---- conv path (channel-last throughout) ----
  {
    int tot = BB * LL * DM;
    k_dwconv3x3<<<(tot + TPB - 1) / TPB, TPB, 0, stream>>>(hs, dw1w, dw1b, t1, DM);
    dim3 g1((BB * LL) / 16, (DM + 63) / 64);
    k_gemm_wmma<1><<<g1, 128, 0, stream>>>(t1, pw1w, pw1b, t2, DM, DM);    // SiLU
    k_dwconv3x3<<<(tot + TPB - 1) / TPB, TPB, 0, stream>>>(t2, dw2w, dw2b, t3, DM);
    dim3 g2((BB * LL) / 16, (DI + 63) / 64);
    k_gemm_wmma<0><<<g2, 128, 0, stream>>>(t3, pw2w, pw2b, ocv, DI, DM);
  }
  // ---- in_proj: xz[b,l,e] = sum_d hs[b,l,d] * ipw[e,d] ----
  {
    dim3 g((BB * LL) / 16, (D2 + 63) / 64);
    k_gemm_wmma<0><<<g, 128, 0, stream>>>(hs, ipw, nullptr, xz, D2, DM);
  }
  // ---- six scan branches (reuse scratch; serialized on stream) ----
  struct Cfg { int base, mode, rev, C, Lb, S, CL; };
  const Cfg br[6] = {
    {10, 0, 0, DI, LL,  8,  128}, {17, 0, 1, DI, LL,  8,  128},
    {24, 1, 0, CI, LCH, 48, 256}, {31, 1, 1, CI, LCH, 48, 256},
    {38, 2, 0, CI, LCH, 48, 256}, {45, 2, 1, CI, LCH, 48, 256},
  };
  for (int i = 0; i < 6; ++i) {
    const Cfg& g = br[i];
    const float* cw   = (const float*)d_in[g.base + 0];
    const float* cb   = (const float*)d_in[g.base + 1];
    const float* xpw  = (const float*)d_in[g.base + 2];
    const float* dtw  = (const float*)d_in[g.base + 3];
    const float* dtb  = (const float*)d_in[g.base + 4];
    const float* alog = (const float*)d_in[g.base + 5];
    const float* Dw   = (const float*)d_in[g.base + 6];

    size_t totc = (size_t)BB * g.Lb * g.C;
    k_causal_conv<<<(unsigned)((totc + TPB - 1) / TPB), TPB, 0, stream>>>(
        xz, cw, cb, xbuf, g.mode, g.rev, g.C, g.Lb);

    dim3 gg((unsigned)((BB * (size_t)g.Lb) / 16), 1);   // N=38 fits one 64-wide block
    k_gemm_wmma<0><<<gg, 128, 0, stream>>>(xbuf, xpw, nullptr, xdb, NJ, g.C);

    int nA = BB * g.C * g.S;
    k_scan_chunkA<<<(nA + TPB - 1) / TPB, TPB, 0, stream>>>(
        xbuf, xdb, alog, dtw, dtb, cbuf, g.C, g.Lb, g.S, g.CL);
    int nB = BB * g.C;
    k_scan_stitch<<<(nB + TPB - 1) / TPB, TPB, 0, stream>>>(cbuf, alog, hinit, g.C, g.S);
    k_scan_chunkC<<<(nA + TPB - 1) / TPB, TPB, 0, stream>>>(
        xbuf, xdb, xz, alog, Dw, dtw, dtb, hinit, acc,
        g.mode, g.rev, g.C, g.Lb, g.S, g.CL);
  }
  // ---- finalize ----
  {
    int tot = BB * LL * DI;
    k_finalize<<<(tot + TPB - 1) / TPB, TPB, 0, stream>>>(acc, ocv, out);
  }
}
